// HierarchicalSparseAttention_77902116815198
// MI455X (gfx1250) — compile-verified, matrix operand order checked by reading, not run on hardware
//
#include <hip/hip_runtime.h>
#include <hip/hip_bf16.h>

// ---------------- problem constants ----------------
#define NB   4096      // N tokens
#define DD   1024      // model dim
#define HH   16        // heads
#define DHd  64        // head dim
#define BBn  2         // batch
#define YROWS 4095     // hierarchy nodes per batch
#define NNODE 8191     // x(4096) + y(4095)
#define SCALE_ 0.125f  // 1/sqrt(64)
#define LVLS 12

typedef __attribute__((ext_vector_type(16))) __bf16 v16bf;
typedef __attribute__((ext_vector_type(8)))  __bf16 v8bf;
typedef __attribute__((ext_vector_type(8)))  float  v8f;

// ---------------- average-pool one level ----------------
__global__ void pool_halve(const float* __restrict__ src, long sbs,
                           float* __restrict__ dst, long dbs,
                           int rows, int batches)
{
    long total = (long)batches * rows * DD;
    for (long i = (long)blockIdx.x * blockDim.x + threadIdx.x; i < total;
         i += (long)gridDim.x * blockDim.x) {
        int d = (int)(i % DD);
        long t = i / DD;
        int r = (int)(t % rows);
        int b = (int)(t / rows);
        const float* s = src + (long)b * sbs + (long)(2 * r) * DD + d;
        dst[(long)b * dbs + (long)r * DD + d] = 0.5f * (s[0] + s[DD]);
    }
}

// ---------------- f32 -> bf16 with per-batch row remap ----------------
__global__ void cvt_bf16(const float* __restrict__ src, __bf16* __restrict__ dst,
                         long count, long rpb, long obs)
{
    for (long i = (long)blockIdx.x * blockDim.x + threadIdx.x; i < count;
         i += (long)gridDim.x * blockDim.x) {
        long row = i / DD;
        int  col = (int)(i % DD);
        long orow = (row / rpb) * obs + (row % rpb);
        dst[orow * DD + col] = (__bf16)src[i];
    }
}

// ---------------- bf16 WMMA GEMM: C[M,N] = A[M,K] * W[N,K]^T (+bias) ----------------
// Block tile 128x128, 8 waves, wave tile 32x64 (2 A-frags x 4 B-frags = 8 WMMA/K-step).
// A rows remapped: arow = (m/a_rpb)*a_bs + a_off + m%a_rpb   (per-batch slicing)
// optional f32 out (plain m), optional bf16 out with its own remap.
#define LDS_P 40   // padded halfword stride (80B) -> conflict-free b128 frag loads
__launch_bounds__(256)
__global__ void gemm_bf16_wmma(const __bf16* __restrict__ A,
                               long a_rpb, long a_bs, long a_off,
                               const __bf16* __restrict__ W,
                               const float* __restrict__ bias,
                               float* __restrict__ Cf,
                               __bf16* __restrict__ Cb,
                               long c_rpb, long c_bs, long c_off,
                               int M, int N, int K)
{
    __shared__ __align__(16) __bf16 As[128 * LDS_P];
    __shared__ __align__(16) __bf16 Bs[128 * LDS_P];

    const int tid  = threadIdx.x;
    const int aRow = tid >> 1, aCol = (tid & 1) * 16;  // 128x32 tile, 16 bf16/thread
    const int bRow = tid >> 1, bCol = (tid & 1) * 16;  // 128x32 tile, 16 bf16/thread

    long gm = (long)blockIdx.x * 128 + aRow; if (gm >= M) gm = M - 1;
    long am = (gm / a_rpb) * a_bs + a_off + (gm % a_rpb);
    const __bf16* aSrc = A + am * (long)K + aCol;
    long gn = (long)blockIdx.y * 128 + bRow; if (gn >= N) gn = N - 1;
    const __bf16* wSrc = W + gn * (long)K + bCol;

    const int lane = tid & 31, wv = tid >> 5;
    const int wy = wv >> 1;          // 0..3 : M sub-tile (32 rows)
    const int wx = wv & 1;           // 0..1 : N sub-tile (64 cols)
    const int r  = lane & 15, h = lane >> 4;

    v8f acc[2][4];
#pragma unroll
    for (int i = 0; i < 2; ++i)
#pragma unroll
        for (int j = 0; j < 4; ++j) { v8f z = {}; acc[i][j] = z; }

    for (int kb = 0; kb < K; kb += 32) {
        __syncthreads();
        *(v8bf*)&As[aRow * LDS_P + aCol]     = *(const v8bf*)(aSrc + kb);
        *(v8bf*)&As[aRow * LDS_P + aCol + 8] = *(const v8bf*)(aSrc + kb + 8);
        *(v8bf*)&Bs[bRow * LDS_P + bCol]     = *(const v8bf*)(wSrc + kb);
        *(v8bf*)&Bs[bRow * LDS_P + bCol + 8] = *(const v8bf*)(wSrc + kb + 8);
        if (kb + 32 < K) {                          // CDNA5 global_prefetch_b8
            __builtin_prefetch(aSrc + kb + 32, 0, 1);
            __builtin_prefetch(wSrc + kb + 32, 0, 1);
        }
        __syncthreads();

        // A fragments: lane r holds row M=r; halves split K per ISA layout
        union { v16bf v; v8bf p[2]; } ua[2];
#pragma unroll
        for (int i = 0; i < 2; ++i) {
            const __bf16* ab = &As[(wy * 32 + i * 16 + r) * LDS_P];
            ua[i].p[0] = *(const v8bf*)(ab + h * 8);
            ua[i].p[1] = *(const v8bf*)(ab + 16 + h * 8);
        }
#pragma unroll
        for (int j = 0; j < 4; ++j) {
            union { v16bf v; v8bf p[2]; } ub;
            const __bf16* bb = &Bs[(wx * 64 + j * 16 + r) * LDS_P];
            ub.p[0] = *(const v8bf*)(bb + h * 8);
            ub.p[1] = *(const v8bf*)(bb + 16 + h * 8);
            acc[0][j] = __builtin_amdgcn_wmma_f32_16x16x32_bf16(
                            false, ua[0].v, false, ub.v, (short)0, acc[0][j], false, false);
            acc[1][j] = __builtin_amdgcn_wmma_f32_16x16x32_bf16(
                            false, ua[1].v, false, ub.v, (short)0, acc[1][j], false, false);
        }
    }

#pragma unroll
    for (int i = 0; i < 2; ++i) {
#pragma unroll
        for (int j = 0; j < 4; ++j) {
#pragma unroll
            for (int v = 0; v < 8; ++v) {
                long m  = (long)blockIdx.x * 128 + wy * 32 + i * 16 + h * 8 + v;
                int col = blockIdx.y * 128 + wx * 64 + j * 16 + r;
                if (m < M && col < N) {
                    float val = acc[i][j][v];
                    if (bias) val += bias[col];
                    if (Cf) Cf[m * (long)N + col] = val;
                    if (Cb) {
                        long cr = (m / c_rpb) * c_bs + c_off + (m % c_rpb);
                        Cb[cr * (long)N + col] = (__bf16)val;
                    }
                }
            }
        }
    }
}

// ---------------- per-level 3-way attention (parent + 2 children) ----------------
__global__ void level_attn(const __bf16* __restrict__ Qp, const __bf16* __restrict__ Kp,
                           const __bf16* __restrict__ Vp, const __bf16* __restrict__ Kc,
                           const __bf16* __restrict__ Vc, __bf16* __restrict__ ao,
                           int pc, int off)
{
    int idx = blockIdx.x * blockDim.x + threadIdx.x;
    int total = BBn * HH * pc;
    if (idx >= total) return;
    int h = idx % HH; int t = idx / HH; int p = t % pc; int b = t / pc;
    long rp  = ((long)b * YROWS + off + p) * DD + h * DHd;
    long rc0 = ((long)b * 2 * pc + 2 * p) * DD + h * DHd;
    long rc1 = rc0 + DD;

    float q[DHd];
#pragma unroll
    for (int d = 0; d < DHd; ++d) q[d] = (float)Qp[rp + d];
    float l0 = 0.f, l1 = 0.f, l2 = 0.f;
#pragma unroll 8
    for (int d = 0; d < DHd; ++d) {
        l0 += q[d] * (float)Kp[rp + d];
        l1 += q[d] * (float)Kc[rc0 + d];
        l2 += q[d] * (float)Kc[rc1 + d];
    }
    l0 *= SCALE_; l1 *= SCALE_; l2 *= SCALE_;
    float mx = fmaxf(l0, fmaxf(l1, l2));
    float w0 = __expf(l0 - mx), w1 = __expf(l1 - mx), w2 = __expf(l2 - mx);
    float inv = 1.f / (w0 + w1 + w2);
    w0 *= inv; w1 *= inv; w2 *= inv;
#pragma unroll 8
    for (int d = 0; d < DHd; ++d) {
        float o = w0 * (float)Vp[rp + d] + w1 * (float)Vc[rc0 + d] + w2 * (float)Vc[rc1 + d];
        ao[rp + d] = (__bf16)o;
    }
}

// ---------------- final tree attention: self + 12 ancestors/siblings ----------------
__global__ void tree_attn(const __bf16* __restrict__ Q, const __bf16* __restrict__ Kf,
                          const __bf16* __restrict__ Vf, __bf16* __restrict__ O)
{
    int idx = blockIdx.x * blockDim.x + threadIdx.x;
    if (idx >= BBn * HH * NB) return;
    int h = idx % HH; int t = idx / HH; int n = t % NB; int b = t / NB;
    long qb = ((long)b * NB + n) * DD + h * DHd;

    float q[DHd];
#pragma unroll
    for (int d = 0; d < DHd; ++d) q[d] = (float)Q[qb + d];

    // rebuild FIDX / FMASK (same recurrence as reference)
    int ids[13]; bool mk[13];
    ids[0] = n; mk[0] = false;
    int ncur = n;
#pragma unroll
    for (int l = 0; l < LVLS; ++l) {
        int pair, nn;
        if (l == 0) { pair = n; nn = n ^ 1; }
        else        { pair = (ncur >> 1) + NB; nn = pair ^ 1; }
        bool valid = nn <= (2 * NB - 3);
        ids[l + 1] = valid ? nn : (2 * NB - 2);
        mk[l + 1]  = valid && (pair < nn);
        ncur = nn;
    }

    float lg[13];
    for (int k = 0; k < 13; ++k) {
        if (mk[k]) { lg[k] = -__builtin_inff(); continue; }
        long kb = ((long)b * NNODE + ids[k]) * DD + h * DHd;
        float s = 0.f;
#pragma unroll 8
        for (int d = 0; d < DHd; ++d) s += q[d] * (float)Kf[kb + d];
        lg[k] = s * SCALE_;
    }
    float mx = lg[0];
    for (int k = 1; k < 13; ++k) mx = fmaxf(mx, lg[k]);
    float sum = 0.f, w[13];
    for (int k = 0; k < 13; ++k) { w[k] = __expf(lg[k] - mx); sum += w[k]; }
    float inv = 1.f / sum;

    float acc[DHd];
#pragma unroll
    for (int d = 0; d < DHd; ++d) acc[d] = 0.f;
    for (int k = 0; k < 13; ++k) {
        if (mk[k]) continue;
        float wk = w[k] * inv;
        long vb = ((long)b * NNODE + ids[k]) * DD + h * DHd;
#pragma unroll 8
        for (int d = 0; d < DHd; ++d) acc[d] += wk * (float)Vf[vb + d];
    }
#pragma unroll
    for (int d = 0; d < DHd; ++d) O[qb + d] = (__bf16)acc[d];
}

// ---------------- host side ----------------
static const long PLAIN = 1LL << 40;

static void launch_gemm(hipStream_t s,
                        const __bf16* A, long arpb, long abs_, long aoff,
                        const __bf16* W, const float* bias,
                        float* Cf, __bf16* Cb, long crpb, long cbs, long coff,
                        int M, int N, int K)
{
    dim3 grid((M + 127) / 128, N / 128);
    gemm_bf16_wmma<<<grid, 256, 0, s>>>(A, arpb, abs_, aoff, W, bias,
                                        Cf, Cb, crpb, cbs, coff, M, N, K);
}

extern "C" void kernel_launch(void* const* d_in, const int* in_sizes, int n_in,
                              void* d_out, int out_size, void* d_ws, size_t ws_size,
                              hipStream_t stream)
{
    const float* x    = (const float*)d_in[0];
    const float* Wq_y = (const float*)d_in[1];
    const float* Wk_y = (const float*)d_in[2];
    const float* Wv_y = (const float*)d_in[3];
    const float* Wo_y = (const float*)d_in[4];
    const float* bo_y = (const float*)d_in[5];
    const float* Wq_x = (const float*)d_in[6];
    const float* Wk_x = (const float*)d_in[7];
    const float* Wv_x = (const float*)d_in[8];
    const float* Wo_x = (const float*)d_in[9];
    const float* bo_x = (const float*)d_in[10];

    // workspace bump allocator
    char* p = (char*)d_ws;
    auto alloc = [&](size_t bytes) -> char* {
        char* r = p; p += (bytes + 255) & ~(size_t)255; return r;
    };
    float*  yin_f = (float*) alloc((size_t)BBn * YROWS * DD * 4);
    __bf16* yin_b = (__bf16*)alloc((size_t)BBn * YROWS * DD * 2);
    __bf16* xy_b  = (__bf16*)alloc((size_t)BBn * NNODE * DD * 2);  // [x ; y] bf16
    __bf16* qp_b  = (__bf16*)alloc((size_t)BBn * YROWS * DD * 2);
    __bf16* kp_b  = (__bf16*)alloc((size_t)BBn * YROWS * DD * 2);
    __bf16* vp_b  = (__bf16*)alloc((size_t)BBn * YROWS * DD * 2);
    __bf16* kc_b  = (__bf16*)alloc((size_t)BBn * NB * DD * 2);
    __bf16* vc_b  = (__bf16*)alloc((size_t)BBn * NB * DD * 2);
    __bf16* ao_b  = (__bf16*)alloc((size_t)BBn * YROWS * DD * 2);
    __bf16* q_b   = (__bf16*)alloc((size_t)BBn * NB * DD * 2);
    __bf16* kf_b  = (__bf16*)alloc((size_t)BBn * NNODE * DD * 2);
    __bf16* vf_b  = (__bf16*)alloc((size_t)BBn * NNODE * DD * 2);
    __bf16* o_b   = (__bf16*)alloc((size_t)BBn * NB * DD * 2);
    __bf16* wb[8];
    for (int i = 0; i < 8; ++i) wb[i] = (__bf16*)alloc((size_t)DD * DD * 2);
    const float* wsrc[8] = {Wq_y, Wk_y, Wv_y, Wo_y, Wq_x, Wk_x, Wv_x, Wo_x};

    int sizes[LVLS], offs[LVLS];
    for (int i = 0; i < LVLS; ++i) sizes[i] = NB >> (i + 1);
    offs[0] = 0;
    for (int i = 1; i < LVLS; ++i) offs[i] = offs[i - 1] + sizes[i - 1];

    const int TPB = 256;
    auto blocks = [](long n) { long b = (n + 255) / 256; return (unsigned)(b > 65535 ? 65535 : b); };

    // 0) converts: x -> XY bf16 (batch stride 8191), weights -> bf16
    cvt_bf16<<<blocks((long)BBn * NB * DD), TPB, 0, stream>>>(
        x, xy_b, (long)BBn * NB * DD, NB, NNODE);
    for (int i = 0; i < 8; ++i)
        cvt_bf16<<<blocks((long)DD * DD), TPB, 0, stream>>>(
            wsrc[i], wb[i], (long)DD * DD, PLAIN, 0);

    // 1) build pooled hierarchy y_in (f32), then convert to bf16
    for (int l = 0; l < LVLS; ++l) {
        int rows = sizes[l];
        const float* src = (l == 0) ? x : (yin_f + (long)offs[l - 1] * DD);
        long sbs = (l == 0) ? (long)NB * DD : (long)YROWS * DD;
        pool_halve<<<blocks((long)BBn * rows * DD), TPB, 0, stream>>>(
            src, sbs, yin_f + (long)offs[l] * DD, (long)YROWS * DD, rows, BBn);
    }
    cvt_bf16<<<blocks((long)BBn * YROWS * DD), TPB, 0, stream>>>(
        yin_f, yin_b, (long)BBn * YROWS * DD, PLAIN, 0);

    // 2) parent projections on y_in (M = 2*4095)
    const int Mp = BBn * YROWS;
    launch_gemm(stream, yin_b, PLAIN, 0, 0, wb[0], nullptr, nullptr, qp_b, PLAIN, 0, 0, Mp, DD, DD);
    launch_gemm(stream, yin_b, PLAIN, 0, 0, wb[1], nullptr, nullptr, kp_b, PLAIN, 0, 0, Mp, DD, DD);
    launch_gemm(stream, yin_b, PLAIN, 0, 0, wb[2], nullptr, nullptr, vp_b, PLAIN, 0, 0, Mp, DD, DD);

    // 3) level-by-level: child K/V projections + 3-way attention
    for (int l = 0; l < LVLS; ++l) {
        int pc = sizes[l], ch = 2 * pc;
        const __bf16* chSrc = (l == 0) ? xy_b : ao_b;
        long arpb = (l == 0) ? NB : ch;
        long abs_ = (l == 0) ? NNODE : YROWS;
        long aoff = (l == 0) ? 0 : offs[l - 1];
        int Mc = BBn * ch;
        launch_gemm(stream, chSrc, arpb, abs_, aoff, wb[1], nullptr, nullptr,
                    kc_b, PLAIN, 0, 0, Mc, DD, DD);
        launch_gemm(stream, chSrc, arpb, abs_, aoff, wb[2], nullptr, nullptr,
                    vc_b, PLAIN, 0, 0, Mc, DD, DD);
        level_attn<<<blocks((long)BBn * HH * pc), TPB, 0, stream>>>(
            qp_b, kp_b, vp_b, kc_b, vc_b, ao_b, pc, offs[l]);
    }

    // 4) y = ao @ Wo_y^T + bo_y, written as bf16 into XY rows [4096, 8191)
    launch_gemm(stream, ao_b, PLAIN, 0, 0, wb[3], bo_y, nullptr,
                xy_b, YROWS, NNODE, NB, Mp, DD, DD);

    // 5) main projections
    launch_gemm(stream, xy_b, NB, NNODE, 0, wb[4], nullptr, nullptr,
                q_b, PLAIN, 0, 0, BBn * NB, DD, DD);
    launch_gemm(stream, xy_b, PLAIN, 0, 0, wb[5], nullptr, nullptr,
                kf_b, PLAIN, 0, 0, BBn * NNODE, DD, DD);
    launch_gemm(stream, xy_b, PLAIN, 0, 0, wb[6], nullptr, nullptr,
                vf_b, PLAIN, 0, 0, BBn * NNODE, DD, DD);

    // 6) tree attention (self + 12 hierarchy neighbors)
    tree_attn<<<blocks((long)BBn * HH * NB), TPB, 0, stream>>>(q_b, kf_b, vf_b, o_b);

    // 7) final output projection (f32 + bias) -> d_out
    launch_gemm(stream, o_b, PLAIN, 0, 0, wb[7], bo_x, (float*)d_out,
                nullptr, PLAIN, 0, 0, BBn * NB, DD, DD);
}
